// GatedGenerator_65077344469481
// MI455X (gfx1250) — matrix operand order repaired and checked by reading, not compiled, not run on hardware
//
#include <hip/hip_runtime.h>

// =====================================================================
// Contextual attention (DeepFill) for MI455X / gfx1250.
//
// ~215 GFLOP vs <1 GB HBM traffic -> matrix-core bound. Both heavy
// stages run on v_wmma_f32_16x16x32_bf16 (wave32, f32 accumulate).
// Tiles are staged with CDNA5 async global->LDS DMA
// (global_load_async_to_lds_b128/b64 + s_wait_asynccnt), double
// buffered so tile k+1 streams in while tile k feeds the WMMA pipe.
//
// Workspace layout (assumes ws_size >= ~116 MB):
//   b_pm / f_pm : padded position-major bf16 [66*66][128]
//   Y           : 4096x4096 f32 correlation logits (64 MB)
//   attnT       : 4096x4096 bf16, [p][q] (32 MB)
//   A2          : 16 x 128 x 4096 bf16 tap matrices (16 MB)
// =====================================================================

typedef __bf16 bf16;
typedef __attribute__((ext_vector_type(16))) __bf16 v16bf;
typedef __attribute__((ext_vector_type(8)))  float  v8f;

#define C_DIM 128
#define HW    128
#define HD    64
#define NP    4096      // HD*HD
#define PW    66        // padded downsampled width (64 + 1 halo each side)
#define PN    (PW*PW)   // 4356

// workspace byte offsets
#define OFF_BPM  0x000000UL   // 4356*128*2 = 1,115,136 B
#define OFF_FPM  0x120000UL
#define OFF_SSQ  0x240000UL   // 4096 f32
#define OFF_MD   0x244000UL   // 4096 f32 (downsampled mask)
#define OFF_INV  0x248000UL   // 4096 f32
#define OFF_MM   0x24C000UL   // 4096 f32
#define OFF_ZERO 0x250000UL   // 256 B zero page (async-load OOB target)
#define OFF_Y    0x400000UL   // 64 MB f32
#define OFF_ATT  0x4400000UL  // 32 MB bf16
#define OFF_A2   0x6400000UL  // 16 MB bf16
// total required: 0x7400000 (116 MB)

// ---- CDNA5 async global->LDS helpers (no builtin coverage; ISA ch.10) ----
__device__ __forceinline__ unsigned lds_off(const void* p) {
  // LDS aperture: flat addr[31:0] == wave-relative LDS address (ISA 10.2)
  return (unsigned)(uintptr_t)p;
}
__device__ __forceinline__ void async_b128(unsigned lds, const void* g) {
  asm volatile("global_load_async_to_lds_b128 %0, %1, off"
               :: "v"(lds), "v"(g) : "memory");
}
__device__ __forceinline__ void async_b64(unsigned lds, const void* g) {
  asm volatile("global_load_async_to_lds_b64 %0, %1, off"
               :: "v"(lds), "v"(g) : "memory");
}

__global__ void zero_page(float* zp) {
  if (threadIdx.x < 64) zp[threadIdx.x] = 0.f;
}

// ------------------------------------------------------------------
// Build padded position-major bf16 layout: pm[(y+1)*66 + (x+1)][c]
// from src downsampled ::2.  Halo rows/cols are zero.
// ------------------------------------------------------------------
__global__ __launch_bounds__(256)
void prep_pm(const float* __restrict__ src, bf16* __restrict__ pm, int item) {
  int idx = blockIdx.x * 256 + threadIdx.x;        // pr*128 + c
  if (idx >= PN * C_DIM) return;
  int c  = idx & 127;
  int pr = idx >> 7;
  int y = pr / PW - 1, x = pr % PW - 1;
  float v = 0.f;
  if ((unsigned)y < HD && (unsigned)x < HD)
    v = src[(((size_t)item * C_DIM + c) * HW + 2 * y) * HW + 2 * x];
  pm[idx] = (bf16)v;
}

// per-position channel sum-of-squares of b_ds, and downsampled mask
__global__ __launch_bounds__(256)
void prep_stats(const float* __restrict__ b, const float* __restrict__ mask,
                float* __restrict__ ssq, float* __restrict__ md, int item) {
  int pos = blockIdx.x * 256 + threadIdx.x;
  if (pos >= NP) return;
  int y = pos >> 6, x = pos & 63;
  float s = 0.f;
  for (int c = 0; c < C_DIM; ++c) {
    float v = b[(((size_t)item * C_DIM + c) * HW + 2 * y) * HW + 2 * x];
    s += v * v;
  }
  ssq[pos] = s;
  md[pos]  = mask[(size_t)item * HW * HW + (2 * y) * HW + 2 * x];
}

// inverse patch norm (3x3 window, zero pad) + mask gate mm
__global__ __launch_bounds__(256)
void prep_norm(const float* __restrict__ ssq, const float* __restrict__ md,
               float* __restrict__ invn, float* __restrict__ mm) {
  int q = blockIdx.x * 256 + threadIdx.x;
  if (q >= NP) return;
  int qy = q >> 6, qx = q & 63;
  float s = 0.f, ms = 0.f;
  for (int dy = -1; dy <= 1; ++dy)
    for (int dx = -1; dx <= 1; ++dx) {
      int yy = qy + dy, xx = qx + dx;
      if ((unsigned)yy < HD && (unsigned)xx < HD) {
        s  += ssq[yy * HD + xx];
        ms += md[yy * HD + xx];
      }
    }
  invn[q] = rsqrtf(s + 1152.f * 1.0e-4f);   // ESCAPE_NAN * 128*3*3
  mm[q]   = (ms == 0.f) ? 1.f : 0.f;
}

// ------------------------------------------------------------------
// WMMA fragment gathers (bf16 16x16x32, wave32; A and B both contiguous k = 16*hi + e)
// ------------------------------------------------------------------
__device__ __forceinline__ v16bf load_frag_a(const bf16* rowK, int hi) {
  v16bf a;
  const bf16* p0 = rowK + hi * 16;
  const bf16* p1 = rowK + hi * 16 + 8;
#pragma unroll
  for (int e = 0; e < 8; ++e) { a[e] = p0[e]; a[e + 8] = p1[e]; }
  return a;
}
__device__ __forceinline__ v16bf load_frag_b(const bf16* colK, int hi) {
  v16bf bv;
  const bf16* p = colK + hi * 16;
#pragma unroll
  for (int e = 0; e < 16; ++e) bv[e] = p[e];
  return bv;
}

// ------------------------------------------------------------------
// GEMM1: Y[q][p] = invn[q] * sum_{9 shifts x 128 ch} b_pm . f_pm
// WG tile 64(q) x 32(p); 8 waves = 4x2 of 16x16 WMMA tiles.
// Async double-buffered staging: each thread issues exactly one b128
// (A) + one b64 (B) per K-step -> uniform s_wait_asynccnt thresholds.
// ------------------------------------------------------------------
__global__ __launch_bounds__(256)
void gemm1(const bf16* __restrict__ bpm, const bf16* __restrict__ fpm,
           const float* __restrict__ invn, float* __restrict__ Y) {
  __shared__ bf16 At[2][64][32];   // [buf][qx][k]  8 KB
  __shared__ bf16 Bt[2][32][32];   // [buf][n][k]   4 KB
  const int tid  = threadIdx.x;
  const int wave = tid >> 5, lane = tid & 31;
  const int hi = lane >> 4, ln = lane & 15;
  const int wq = wave & 3, wp = wave >> 2;
  const int qy  = blockIdx.y;                 // full q row (64 q's)
  const int pb  = blockIdx.x * 32;
  const int py  = pb >> 6, px0 = pb & 63;

  const int ar = tid >> 2, ako = (tid & 3) * 8;   // A: 64 rows x 32k, 16B/thread
  const int bn = tid >> 3, bko = (tid & 7) * 4;   // B: 32 rows x 32k,  8B/thread

  auto issue = [&](int step, int buf) {
    const int s = step >> 2, cb = step & 3;
    const int dy = s / 3 - 1, dx = s % 3 - 1;
    const bf16* ga = bpm + ((size_t)(qy + dy + 1) * PW + (ar + dx + 1)) * C_DIM
                         + cb * 32 + ako;
    const bf16* gb = fpm + ((size_t)(py + dy + 1) * PW + (px0 + bn + dx + 1)) * C_DIM
                         + cb * 32 + bko;
    async_b128(lds_off(&At[buf][ar][ako]), ga);
    async_b64 (lds_off(&Bt[buf][bn][bko]), gb);
  };

  issue(0, 0);
  v8f acc = {};
  for (int step = 0; step < 36; ++step) {
    const int cur = step & 1;
    if (step + 1 < 36) {
      issue(step + 1, cur ^ 1);
      asm volatile("s_wait_asynccnt 0x2" ::: "memory");  // tile `cur` landed
    } else {
      asm volatile("s_wait_asynccnt 0x0" ::: "memory");
    }
    __syncthreads();
    v16bf a  = load_frag_a(&At[cur][wq * 16 + ln][0], hi);
    v16bf bv = load_frag_b(&Bt[cur][wp * 16 + ln][0], hi);
    acc = __builtin_amdgcn_wmma_f32_16x16x32_bf16(
        false, a, false, bv, (short)0, acc, false, false);
    __syncthreads();   // protect buffer reuse by next-next issue
  }
  const int p = pb + wp * 16 + ln;
#pragma unroll
  for (int r = 0; r < 8; ++r) {
    const int q = qy * 64 + wq * 16 + r + hi * 8;  // C/D: VGPR r -> M=r(+8)
    Y[(size_t)q * NP + p] = acc[r] * invn[q];
  }
}

// flat <-> transposed-flat index map of the 64x64 factorization
__device__ __forceinline__ int Tr(int v) { return ((v & 63) << 6) | (v >> 6); }

// ------------------------------------------------------------------
// Both identity-kernel "fuse" passes (diagonal sums with the original
// code's flat-index wrap semantics) + masked softmax over q.
// One block per p; writes attnT[p][q] bf16 (q contiguous for GEMM2-B).
// ------------------------------------------------------------------
__global__ __launch_bounds__(256)
void fuse_softmax(const float* __restrict__ Y, const float* __restrict__ mm,
                  bf16* __restrict__ att) {
  __shared__ float row[NP];
  __shared__ float red[256];
  const int p = blockIdx.x, tid = threadIdx.x;
  const int j = Tr(p);
  float lmax = -1e30f;
  for (int q = tid; q < NP; q += 256) {
    const int i = Tr(q);
    float F = 0.f;
#pragma unroll
    for (int d2 = -1; d2 <= 1; ++d2) {
      const int a = i + d2, bj = j + d2;
      if ((unsigned)a < NP && (unsigned)bj < NP) {
        const int qa = Tr(a), pv = Tr(bj);
#pragma unroll
        for (int d1 = -1; d1 <= 1; ++d1) {
          const int rr = qa + d1, cc = pv + d1;
          if ((unsigned)rr < NP && (unsigned)cc < NP)
            F += Y[(size_t)rr * NP + cc];
        }
      }
    }
    const float L = F * mm[q];
    row[q] = L;
    lmax = fmaxf(lmax, L);
  }
  red[tid] = lmax; __syncthreads();
  for (int s2 = 128; s2 > 0; s2 >>= 1) {
    if (tid < s2) red[tid] = fmaxf(red[tid], red[tid + s2]);
    __syncthreads();
  }
  const float gmax = red[0];
  __syncthreads();
  float lsum = 0.f;
  for (int q = tid; q < NP; q += 256) {
    const float e = __expf(10.f * (row[q] - gmax));
    row[q] = e;
    lsum += e;
  }
  red[tid] = lsum; __syncthreads();
  for (int s2 = 128; s2 > 0; s2 >>= 1) {
    if (tid < s2) red[tid] += red[tid + s2];
    __syncthreads();
  }
  const float inv = 1.f / red[0];
  for (int q = tid; q < NP; q += 256)
    att[(size_t)p * NP + q] = (bf16)(row[q] * inv * mm[q]);
}

// ------------------------------------------------------------------
// A-matrices for the transposed conv: A2[(cls*4+combo)][c][q] =
// 0.25 * raw_patch[q][c][ry][rx], taps chosen by output parity class.
// ------------------------------------------------------------------
__global__ __launch_bounds__(256)
void prepA(const float* __restrict__ b, bf16* __restrict__ A2, int item) {
  const size_t idx = (size_t)blockIdx.x * 256 + threadIdx.x;
  const int q  = (int)(idx & (NP - 1));
  const int c  = (int)((idx >> 12) & 127);
  const int cc = (int)(idx >> 19);            // 0..15
  const int cls = cc >> 2, combo = cc & 3;
  const int sy = cls >> 1, sx = cls & 1;
  const int iy = combo >> 1, ix = combo & 1;
  const int ry = sy ? (iy ? 0 : 2) : (iy ? 3 : 1);
  const int rx = sx ? (ix ? 0 : 2) : (ix ? 3 : 1);
  const int qy = q >> 6, qx = q & 63;
  const int Yb = 2 * qy + ry - 1, Xb = 2 * qx + rx - 1;
  float v = 0.f;
  if ((unsigned)Yb < HW && (unsigned)Xb < HW)
    v = b[(((size_t)item * C_DIM + c) * HW + Yb) * HW + Xb];
  A2[idx] = (bf16)(0.25f * v);                // fold the /4.0
}

// ------------------------------------------------------------------
// GEMM2 (transposed conv as 4 parity-class GEMMs):
// OutClass[c][pos] = sum_{combo} sum_q A2 . attnT(shifted).
// WG: M=128 (8 waves x 16 ch) x N=32 positions; K = 4 combos x 4096 q.
// Async double-buffered staging: 2x b128 (A) + 1x b64 (B) per thread
// per step; OOB B rows redirected to a zero page.
// ------------------------------------------------------------------
__global__ __launch_bounds__(256)
void gemm2(const bf16* __restrict__ A2, const bf16* __restrict__ att,
           const bf16* __restrict__ zp, float* __restrict__ out, int item) {
  __shared__ bf16 At[2][128][32];  // 16 KB
  __shared__ bf16 Bt[2][32][32];   //  4 KB
  const int tid  = threadIdx.x;
  const int wave = tid >> 5, lane = tid & 31;
  const int hi = lane >> 4, ln = lane & 15;
  const int cls = blockIdx.y;
  const int sy = cls >> 1, sx = cls & 1;
  const int nb0 = blockIdx.x * 32;
  const int ty = nb0 >> 6, tx0 = nb0 & 63;

  const int cr = tid >> 1, ako = (tid & 1) * 16;  // A: 128 rows, 32B/thread
  const int bn = tid >> 3, bko = (tid & 7) * 4;   // B: 32 rows,   8B/thread

  auto issue = [&](int step, int buf) {
    const int combo = step >> 7, ks = step & 127;
    const int iy = combo >> 1, ix = combo & 1;
    const int ay = sy ? (iy ? 1 : 0) : (iy ? -1 : 0);
    const int ax = sx ? (ix ? 1 : 0) : (ix ? -1 : 0);
    const int pyy = ty + ay;
    const int pxx = tx0 + bn + ax;
    const bf16* ga = A2 + ((size_t)(cls * 4 + combo) * C_DIM + cr) * NP
                        + ks * 32 + ako;
    const bf16* gb = ((unsigned)pyy < HD && (unsigned)pxx < HD)
                   ? att + (size_t)(pyy * 64 + pxx) * NP + ks * 32 + bko
                   : zp;
    async_b128(lds_off(&At[buf][cr][ako]),     ga);
    async_b128(lds_off(&At[buf][cr][ako + 8]), ga + 8);
    async_b64 (lds_off(&Bt[buf][bn][bko]),     gb);
  };

  issue(0, 0);
  v8f acc0 = {}, acc1 = {};
  for (int step = 0; step < 512; ++step) {
    const int cur = step & 1;
    if (step + 1 < 512) {
      issue(step + 1, cur ^ 1);
      asm volatile("s_wait_asynccnt 0x3" ::: "memory");  // tile `cur` landed
    } else {
      asm volatile("s_wait_asynccnt 0x0" ::: "memory");
    }
    __syncthreads();
    v16bf a  = load_frag_a(&At[cur][wave * 16 + ln][0], hi);
    v16bf b0 = load_frag_b(&Bt[cur][ln][0], hi);
    v16bf b1 = load_frag_b(&Bt[cur][16 + ln][0], hi);
    acc0 = __builtin_amdgcn_wmma_f32_16x16x32_bf16(
        false, a, false, b0, (short)0, acc0, false, false);
    acc1 = __builtin_amdgcn_wmma_f32_16x16x32_bf16(
        false, a, false, b1, (short)0, acc1, false, false);
    __syncthreads();
  }
#pragma unroll
  for (int r = 0; r < 8; ++r) {
    const int c = wave * 16 + r + hi * 8;
    {
      const int nb = nb0 + ln;
      const int tyv = nb >> 6, txv = nb & 63;
      out[(((size_t)item * C_DIM + c) * HW + (2 * tyv + sy)) * HW + (2 * txv + sx)] = acc0[r];
    }
    {
      const int nb = nb0 + 16 + ln;
      const int tyv = nb >> 6, txv = nb & 63;
      out[(((size_t)item * C_DIM + c) * HW + (2 * tyv + sy)) * HW + (2 * txv + sx)] = acc1[r];
    }
  }
}

// ------------------------------------------------------------------
extern "C" void kernel_launch(void* const* d_in, const int* in_sizes, int n_in,
                              void* d_out, int out_size, void* d_ws, size_t ws_size,
                              hipStream_t stream) {
  const float* f    = (const float*)d_in[0];
  const float* b    = (const float*)d_in[1];
  const float* mask = (const float*)d_in[2];
  float* out = (float*)d_out;
  char*  ws  = (char*)d_ws;

  bf16*  bpm  = (bf16*) (ws + OFF_BPM);
  bf16*  fpm  = (bf16*) (ws + OFF_FPM);
  float* ssq  = (float*)(ws + OFF_SSQ);
  float* md   = (float*)(ws + OFF_MD);
  float* invn = (float*)(ws + OFF_INV);
  float* mmv  = (float*)(ws + OFF_MM);
  float* zp   = (float*)(ws + OFF_ZERO);
  float* Yb   = (float*)(ws + OFF_Y);
  bf16*  att  = (bf16*) (ws + OFF_ATT);
  bf16*  A2   = (bf16*) (ws + OFF_A2);

  const int pm_grid = (PN * C_DIM + 255) / 256;   // 2178

  zero_page<<<1, 64, 0, stream>>>(zp);
  for (int item = 0; item < 2; ++item) {
    prep_pm   <<<pm_grid, 256, 0, stream>>>(b, bpm, item);
    prep_pm   <<<pm_grid, 256, 0, stream>>>(f, fpm, item);
    prep_stats<<<NP / 256, 256, 0, stream>>>(b, mask, ssq, md, item);
    prep_norm <<<NP / 256, 256, 0, stream>>>(ssq, md, invn, mmv);
    gemm1     <<<dim3(128, 64), 256, 0, stream>>>(bpm, fpm, invn, Yb);
    fuse_softmax<<<NP, 256, 0, stream>>>(Yb, mmv, att);
    prepA     <<<(16 * C_DIM * NP) / 256, 256, 0, stream>>>(b, A2, item);
    gemm2     <<<dim3(128, 4), 256, 0, stream>>>(A2, att, (const bf16*)zp, out, item);
  }
}
